// OutputLayer_KAN_1layer_3289944949182
// MI455X (gfx1250) — matrix operand order repaired and checked by reading, not compiled, not run on hardware
//
#include <hip/hip_runtime.h>
#include <hip/hip_bf16.h>
#include <math.h>

typedef __attribute__((ext_vector_type(16))) __bf16 v16bf;
typedef __attribute__((ext_vector_type(8)))  float  v8f;

#define B_ROWS   16384
#define D_IN     512
#define D_HID    256
#define NBASIS   8
#define KSPLINE  (D_IN * NBASIS)   // 4096
#define NCLS     8
#define ATTN     9
#define DCOMB    (D_HID + ATTN)    // 265
#define XS_PITCH 516               // 512 + 4 floats pad (breaks stride-512 bank conflicts)

// ---------------- ws layout ----------------
// [0,               2*256*4096) : bf16 scaled spline weight  [256][4096]
// [2*256*4096, +2*256*512)      : bf16 base weight           [256][512]

// Branch-free erf (Abramowitz–Stegun 7.1.26, |err| < 1.5e-7) -> branch-free gelu.
__device__ __forceinline__ float gelu_fast(float x) {
    const float z  = x * 0.70710678118654752f;       // x / sqrt(2)
    const float az = fabsf(z);
    const float t  = __fdividef(1.0f, fmaf(0.3275911f, az, 1.0f));
    float p = fmaf(1.061405429f, t, -1.453152027f);
    p = fmaf(p, t,  1.421413741f);
    p = fmaf(p, t, -0.284496736f);
    p = fmaf(p, t,  0.254829592f);
    p = p * t;
    const float e   = __expf(-az * az);
    const float erf = copysignf(fmaf(-p, e, 1.0f), z);
    return 0.5f * x * (1.0f + erf);
}

// Uniform-knot cubic B-spline basis: 8 values for one x.
// knots t[i] = -1 + (i-3)*0.4, i = 0..11
__device__ __forceinline__ void spline8(float x, float out[8]) {
    const float h = 0.4f;
    float t[12];
#pragma unroll
    for (int i = 0; i < 12; ++i) t[i] = -1.0f + (float)(i - 3) * h;
    float b[11];
#pragma unroll
    for (int j = 0; j < 11; ++j) b[j] = (x >= t[j] && x < t[j + 1]) ? 1.0f : 0.0f;
#pragma unroll
    for (int d = 1; d <= 3; ++d) {
        const float inv = 1.0f / ((float)d * h);   // constant-folded
#pragma unroll
        for (int j = 0; j < 10; ++j) {
            if (j < 11 - d) {
                float left  = (x - t[j]) * inv;
                float right = (t[j + d + 1] - x) * inv;
                b[j] = left * b[j] + right * b[j + 1];
            }
        }
    }
#pragma unroll
    for (int j = 0; j < 8; ++j) out[j] = b[j];
}

// ---------------- kernel 1: weight prep (f32 -> bf16, fold scaler) -------------
__global__ __launch_bounds__(256) void prep_weights(
        const float* __restrict__ spline_w,   // [256][512][8]
        const float* __restrict__ scaler,     // [256][512]
        const float* __restrict__ base_w,     // [256][512]
        __bf16* __restrict__ sw_bf,           // [256][4096]
        __bf16* __restrict__ bw_bf)           // [256][512]
{
    int idx = blockIdx.x * blockDim.x + threadIdx.x;
    if (idx < D_HID * KSPLINE)
        sw_bf[idx] = (__bf16)(spline_w[idx] * scaler[idx >> 3]);
    if (idx < D_HID * D_IN)
        bw_bf[idx] = (__bf16)base_w[idx];
}

// ---------------- kernel 2: fused KAN layer via bf16 WMMA ----------------------
// Block: 128 threads (4 waves). Wave w handles rows [rowBlk + 16w, +16),
// cols [nBase, +128) as 8 WMMA n-tiles. sp is generated entirely in registers.
__global__ __launch_bounds__(128) void kan_fused(
        const float*  __restrict__ xg,        // [16384][512]
        const float*  __restrict__ attn,      // [16384][9]
        const __bf16* __restrict__ sw_bf,     // [256][4096]
        const __bf16* __restrict__ bw_bf,     // [256][512]
        float* __restrict__ out_comb)         // [16384][265]
{
    extern __shared__ float xs[];             // [64][XS_PITCH]
    const int tid    = threadIdx.x;
    const int lane   = tid & 31;
    const int wave   = tid >> 5;
    const int rowBlk = blockIdx.x * 64;
    const int nBase  = blockIdx.y * 128;

    // ---- stage 64 rows of x into LDS (float4, coalesced) ----
    const float4* xg4 = (const float4*)(xg + (size_t)rowBlk * D_IN);
    for (int i = tid; i < 64 * (D_IN / 4); i += 128) {
        int r  = i >> 7;       // / (D_IN/4)
        int c4 = i & 127;
        float4 v = xg4[r * (D_IN / 4) + c4];
        *(float4*)(xs + r * XS_PITCH + c4 * 4) = v;
    }
    __syncthreads();

    const int hi   = lane >> 4;          // lane group (0: lanes 0-15, 1: 16-31)
    const int ncol = lane & 15;
    const int aoff = hi ? 16 : 0;        // A-fragment K offset for this lane group
    const int boff = hi ? 16 : 0;        // B-fragment K offset for this lane group
    const int lrow = wave * 16 + ncol;   // this lane's M row within the LDS tile
    const float* xrow = xs + lrow * XS_PITCH;

    // Per-lane base pointers for the 8 B-fragment rows (n = nBase + t*16 + ncol)
    const __bf16* bwp[8];
    const __bf16* swp[8];
#pragma unroll
    for (int t = 0; t < 8; ++t) {
        const int n = nBase + t * 16 + ncol;
        bwp[t] = bw_bf + (size_t)n * D_IN    + boff;
        swp[t] = sw_bf + (size_t)n * KSPLINE + boff;
    }

    v8f acc[8] = {};

    // ---- phase 1: base_out = gelu(x) @ base_w^T  (K = 512) ----
    for (int kb = 0; kb < D_IN; kb += 32) {
        // 1) issue all 8 B-fragment loads (one clause, one wait point)
        v16bf bf[8];
#pragma unroll
        for (int t = 0; t < 8; ++t) bf[t] = *(const v16bf*)(bwp[t] + kb);
        // prefetch next chunk's weight lines into cache (global_prefetch_b8)
        if (kb + 32 < D_IN) {
#pragma unroll
            for (int t = 0; t < 8; ++t) __builtin_prefetch(bwp[t] + kb + 32, 0, 1);
        }
        // 2) build A fragment from LDS while loads are in flight
        const int k0 = kb + aoff;
        v16bf a;
#pragma unroll
        for (int j = 0; j < 8; ++j) a[j]     = (__bf16)gelu_fast(xrow[k0 + j]);
#pragma unroll
        for (int j = 0; j < 8; ++j) a[8 + j] = (__bf16)gelu_fast(xrow[k0 + 8 + j]);
        // 3) WMMA burst
#pragma unroll
        for (int t = 0; t < 8; ++t)
            acc[t] = __builtin_amdgcn_wmma_f32_16x16x32_bf16(
                         false, a, false, bf[t], (short)0, acc[t], false, false);
    }

    // ---- phase 2: h += b_splines(x) @ scaled_w^T  (K = 4096, sp in-register) ----
    for (int kb = 0; kb < KSPLINE; kb += 32) {
        v16bf bf[8];
#pragma unroll
        for (int t = 0; t < 8; ++t) bf[t] = *(const v16bf*)(swp[t] + kb);
        if (kb + 32 < KSPLINE) {
#pragma unroll
            for (int t = 0; t < 8; ++t) __builtin_prefetch(swp[t] + kb + 32, 0, 1);
        }
        const int k0 = kb + aoff;
        const int fA = k0 >> 3;           // this lane's two features: fA, fA+1
        float bA[8], bB[8];
        spline8(xrow[fA],     bA);
        spline8(xrow[fA + 1], bB);
        v16bf a;
#pragma unroll
        for (int j = 0; j < 8; ++j) { a[j] = (__bf16)bA[j]; a[8 + j] = (__bf16)bB[j]; }
#pragma unroll
        for (int t = 0; t < 8; ++t)
            acc[t] = __builtin_amdgcn_wmma_f32_16x16x32_bf16(
                         false, a, false, bf[t], (short)0, acc[t], false, false);
    }

    // ---- epilogue: scatter C fragments into combined_x ----
    // C layout: VGPR j, lanes 0-15 -> (M=j, N=lane); lanes 16-31 -> (M=8+j, N=lane-16)
#pragma unroll
    for (int t = 0; t < 8; ++t) {
#pragma unroll
        for (int j = 0; j < 8; ++j) {
            const int m = rowBlk + wave * 16 + (hi ? j + 8 : j);
            const int n = nBase + t * 16 + ncol;
            out_comb[(size_t)m * DCOMB + n] = acc[t][j];
        }
    }
    // append attn_embedding columns (only one n-half does it)
    if (blockIdx.y == 0 && ncol < ATTN) {
#pragma unroll
        for (int j = 0; j < 8; ++j) {
            const int m = rowBlk + wave * 16 + (hi ? j + 8 : j);
            out_comb[(size_t)m * DCOMB + D_HID + ncol] = attn[(size_t)m * ATTN + ncol];
        }
    }
}

// ---------------- kernel 3: classifier head + log_softmax ---------------------
__global__ __launch_bounds__(256) void head_kernel(
        const float* __restrict__ comb,    // [16384][265]
        const float* __restrict__ clsw,    // [8][265]
        const float* __restrict__ clsb,    // [8]
        float* __restrict__ out_ls)        // [16384][8]
{
    int g = blockIdx.x * blockDim.x + threadIdx.x;
    if (g >= B_ROWS * NCLS) return;
    const int row = g >> 3;
    const int c   = g & 7;
    const float* cr = comb + (size_t)row * DCOMB;
    const float* wr = clsw + (size_t)c * DCOMB;
    float dot = clsb[c];
    for (int n = 0; n < DCOMB; ++n) dot += cr[n] * wr[n];
    // log_softmax across the 8 classes that live in one 8-lane group
    float mx = dot;
    for (int s = 1; s < 8; s <<= 1) mx = fmaxf(mx, __shfl_xor(mx, s, 8));
    float e = __expf(dot - mx);
    float sum = e;
    for (int s = 1; s < 8; s <<= 1) sum += __shfl_xor(sum, s, 8);
    out_ls[g] = dot - mx - __logf(sum);
}

extern "C" void kernel_launch(void* const* d_in, const int* in_sizes, int n_in,
                              void* d_out, int out_size, void* d_ws, size_t ws_size,
                              hipStream_t stream) {
    const float* x        = (const float*)d_in[0];
    const float* attn     = (const float*)d_in[1];
    const float* base_w   = (const float*)d_in[2];
    const float* spline_w = (const float*)d_in[3];
    const float* scaler   = (const float*)d_in[4];
    const float* clsw     = (const float*)d_in[5];
    const float* clsb     = (const float*)d_in[6];

    float* out_ls   = (float*)d_out;                              // 16384*8
    float* out_comb = (float*)d_out + (size_t)B_ROWS * NCLS;      // 16384*265

    __bf16* sw_bf = (__bf16*)d_ws;
    __bf16* bw_bf = (__bf16*)((char*)d_ws + (size_t)D_HID * KSPLINE * sizeof(__bf16));

    const int prepN = D_HID * KSPLINE;
    prep_weights<<<(prepN + 255) / 256, 256, 0, stream>>>(
        spline_w, scaler, base_w, sw_bf, bw_bf);

    dim3 grid(B_ROWS / 64, 2);
    const size_t smem = (size_t)64 * XS_PITCH * sizeof(float);    // 132 KB of 320 KB WGP LDS
    kan_fused<<<grid, 128, smem, stream>>>(x, attn, sw_bf, bw_bf, out_comb);

    head_kernel<<<(B_ROWS * NCLS + 255) / 256, 256, 0, stream>>>(
        out_comb, clsw, clsb, out_ls);
}
